// CXLAwareKCustomAttention_24610162606086
// MI455X (gfx1250) — matrix operand order repaired and checked by reading, not compiled, not run on hardware
//
#include <hip/hip_runtime.h>

// CDNA5 (gfx1250) block-sparse attention, bf16 WMMA with f32 accumulation.
// Reference semantics: per-kv-block softmax, summed over blocks (no online max).

typedef __attribute__((ext_vector_type(16))) __bf16 v16bf;
typedef __attribute__((ext_vector_type(8)))  __bf16 bf16x8;
typedef __attribute__((ext_vector_type(8)))  float  v8f;

constexpr int BLOCK   = 128;            // kv / query block size
constexpr int LOCBLKS = 1024 / BLOCK;   // LOCAL_WIN in blocks = 8
constexpr int TOPK    = 16;
constexpr int H       = 16;
constexpr int D       = 128;
constexpr int HIDDEN  = H * D;          // 2048
constexpr int WAVES   = 8;
constexpr int THREADS = WAVES * 32;

// ---- fragment loaders (ISA 7.12.2 layouts, wave32) ----

// A-matrix 16x32 bf16: lane m = lane&15; VGPR v: k = (v&3)*2 + 8*(lane>=16) + 16*(v>=4)
__device__ inline v16bf fragA_lds(const __bf16* __restrict__ base, int ld, int lane) {
  const __bf16* row = base + (lane & 15) * ld;
  const int kh = (lane >> 4) << 3;
  v16bf f;
#pragma unroll
  for (int v = 0; v < 8; ++v) {
    int k = ((v & 3) << 1) + kh + ((v >> 2) << 4);
    f[2 * v]     = row[k];
    f[2 * v + 1] = row[k + 1];
  }
  return f;
}

// A fragment built from global f32 rows (scaled, converted to bf16)
__device__ inline v16bf fragA_f32(const float* __restrict__ base, int ld, float scale, int lane) {
  const float* row = base + (lane & 15) * ld;
  const int kh = (lane >> 4) << 3;
  v16bf f;
#pragma unroll
  for (int v = 0; v < 8; ++v) {
    int k = ((v & 3) << 1) + kh + ((v >> 2) << 4);
    f[2 * v]     = (__bf16)(row[k] * scale);
    f[2 * v + 1] = (__bf16)(row[k + 1] * scale);
  }
  return f;
}

// B-matrix 32x16 bf16, B = X^T with X row-major [n][k]:
// lane n = lane&15; VGPR v: k = (v&3)*2 + 8*(lane>=16) + 16*(v>=4)
__device__ inline v16bf fragB_lds(const __bf16* __restrict__ base, int ld, int lane) {
  const __bf16* row = base + (lane & 15) * ld;
  const int kh = (lane >> 4) << 3;
  v16bf f;
#pragma unroll
  for (int v = 0; v < 8; ++v) {
    int k = ((v & 3) << 1) + kh + ((v >> 2) << 4);
    f[2 * v]     = row[k];
    f[2 * v + 1] = row[k + 1];
  }
  return f;
}

__global__ __launch_bounds__(THREADS, 1)
void sparse_attn_bf16_wmma(const float* __restrict__ query,
                           const float* __restrict__ key,
                           const float* __restrict__ value,
                           const float* __restrict__ cache_k,
                           const float* __restrict__ cache_v,
                           const int*   __restrict__ pos_ptr,
                           float* __restrict__ out)
{
  __shared__ __attribute__((aligned(16))) __bf16 lds_k [BLOCK * D];  // K tile [key][d]; recycled as P
  __shared__ __attribute__((aligned(16))) __bf16 lds_vt[D * BLOCK];  // V tile transposed [d][key]

  const int h    = blockIdx.x;
  const int qblk = blockIdx.y;
  const int tid  = threadIdx.x;
  const int lane = tid & 31;
  const int wave = tid >> 5;
  const int cpos = pos_ptr[0];

  const float scale = rsqrtf((float)D);

  // Q strip for this wave (16 rows x 128 d) as 4 register A-fragments, pre-scaled.
  const float* qbase =
      query + (size_t)(qblk * BLOCK + wave * 16) * HIDDEN + h * D;
  v16bf aq[4];
#pragma unroll
  for (int ks = 0; ks < 4; ++ks)
    aq[ks] = fragA_f32(qbase + ks * 32, HIDDEN, scale, lane);

  // Output accumulators: 8 d-tiles of 16x16 f32.
  v8f acc[8];
#pragma unroll
  for (int t = 0; t < 8; ++t) acc[t] = (v8f){};

  // Attended block set: union of local window and strided "important" blocks.
  const int cur    = (cpos + qblk * BLOCK) / BLOCK;
  const int lo     = (cur >= LOCBLKS) ? cur - LOCBLKS : 0;
  const int total  = (cpos + qblk * BLOCK + BLOCK) / BLOCK;
  int stride = total / TOPK; if (stride < 1) stride = 1;

  for (int b = 0; b <= cur; ++b) {
    if (!((b >= lo) || (b % stride == 0))) continue;   // uniform across block

    // ---- cooperative gather: K/V block -> bf16 LDS, 8 f32 per lane per step ----
    // (branch cache vs freshly-written kv region; never mutates the cache input)
    const int base_pos = b * BLOCK;
#pragma unroll
    for (int it = 0; it < (BLOCK * D / 8) / THREADS; ++it) {   // 8 iterations
      const int idx = tid + it * THREADS;
      const int r   = idx >> 4;          // key row in tile (16 chunks/row)
      const int c8  = (idx & 15) << 3;   // starting d of this 8-float chunk
      const int p   = base_pos + r;
      size_t off;
      const float *kp, *vp;
      if (p >= cpos) {                   // fresh region: read inputs directly
        off = (size_t)(p - cpos) * HIDDEN + (size_t)h * D + c8;
        kp = key;   vp = value;
      } else {
        off = (size_t)p * HIDDEN + (size_t)h * D + c8;
        kp = cache_k; vp = cache_v;
      }
      const float4 k0 = *(const float4*)(kp + off);
      const float4 k1 = *(const float4*)(kp + off + 4);
      const float4 v0 = *(const float4*)(vp + off);
      const float4 v1 = *(const float4*)(vp + off + 4);

      bf16x8 kb;
      kb[0] = (__bf16)k0.x; kb[1] = (__bf16)k0.y;
      kb[2] = (__bf16)k0.z; kb[3] = (__bf16)k0.w;
      kb[4] = (__bf16)k1.x; kb[5] = (__bf16)k1.y;
      kb[6] = (__bf16)k1.z; kb[7] = (__bf16)k1.w;
      *(bf16x8*)(lds_k + r * D + c8) = kb;           // one b128 DS store

      __bf16* vtcol = lds_vt + c8 * BLOCK + r;       // transpose scatter
      vtcol[0 * BLOCK] = (__bf16)v0.x; vtcol[1 * BLOCK] = (__bf16)v0.y;
      vtcol[2 * BLOCK] = (__bf16)v0.z; vtcol[3 * BLOCK] = (__bf16)v0.w;
      vtcol[4 * BLOCK] = (__bf16)v1.x; vtcol[5 * BLOCK] = (__bf16)v1.y;
      vtcol[6 * BLOCK] = (__bf16)v1.z; vtcol[7 * BLOCK] = (__bf16)v1.w;
    }

    // ---- prefetch next attended block's K/V (global_prefetch_b8) ----
    {
      int nb = -1;
      for (int b2 = b + 1; b2 <= cur; ++b2)
        if ((b2 >= lo) || (b2 % stride == 0)) { nb = b2; break; }
      if (nb >= 0) {
        const int r = tid >> 1;                 // 128 rows, 2 lines each
        const int half = (tid & 1) << 6;        // 0 or 64 floats (256B)
        const int p = nb * BLOCK + r;
        size_t off;
        const float *kp, *vp;
        if (p >= cpos) { off = (size_t)(p - cpos) * HIDDEN + (size_t)h * D + half; kp = key;     vp = value;   }
        else           { off = (size_t)p          * HIDDEN + (size_t)h * D + half; kp = cache_k; vp = cache_v; }
        __builtin_prefetch(kp + off, 0, 1);
        __builtin_prefetch(vp + off, 0, 1);
      }
    }
    __syncthreads();

    // ---- scores S = (Q*scale) Kt : 8 key-tiles, processed in pairs so each
    //      ks step has two independent B-load streams + accumulator chains ----
    v8f sc[8];
#pragma unroll
    for (int tp = 0; tp < 4; ++tp) {
      const int t0 = 2 * tp, t1 = 2 * tp + 1;
      v8f c0 = {}, c1 = {};
#pragma unroll
      for (int ks = 0; ks < 4; ++ks) {
        v16bf b0 = fragB_lds(lds_k + t0 * 16 * D + ks * 32, D, lane);
        v16bf b1 = fragB_lds(lds_k + t1 * 16 * D + ks * 32, D, lane);
        c0 = __builtin_amdgcn_wmma_f32_16x16x32_bf16(
                 false, aq[ks], false, b0, (short)0, c0, false, false);
        c1 = __builtin_amdgcn_wmma_f32_16x16x32_bf16(
                 false, aq[ks], false, b1, (short)0, c1, false, false);
      }
      sc[t0] = c0;
      sc[t1] = c1;
    }

    // ---- per-kv-block softmax over 128 keys (row M = j + 8*laneHi) ----
#pragma unroll
    for (int j = 0; j < 8; ++j) {
      float m = sc[0][j];
#pragma unroll
      for (int t = 1; t < 8; ++t) m = fmaxf(m, sc[t][j]);
#pragma unroll
      for (int x = 1; x <= 8; x <<= 1) m = fmaxf(m, __shfl_xor(m, x, 32));
      float s = 0.f;
#pragma unroll
      for (int t = 0; t < 8; ++t) {
        float e = __expf(sc[t][j] - m);
        sc[t][j] = e;
        s += e;
      }
#pragma unroll
      for (int x = 1; x <= 8; x <<= 1) s += __shfl_xor(s, x, 32);
      const float inv = 1.0f / s;
#pragma unroll
      for (int t = 0; t < 8; ++t) sc[t][j] *= inv;
    }

    __syncthreads();   // everyone done reading lds_k; recycle it as P buffer

    // ---- spill P (bf16) into per-wave 16x128 region of recycled lds_k ----
    __bf16* pbuf = lds_k + wave * 16 * BLOCK;
#pragma unroll
    for (int t = 0; t < 8; ++t) {
#pragma unroll
      for (int j = 0; j < 8; ++j) {
        const int M = j + ((lane >> 4) << 3);
        const int N = (lane & 15) + t * 16;
        pbuf[M * BLOCK + N] = (__bf16)sc[t][j];
      }
    }

    // ---- O += P V : A = P chunks, B = Vt rows (8 independent chains per ks) ----
#pragma unroll
    for (int ks = 0; ks < 4; ++ks) {
      v16bf ap = fragA_lds(pbuf + ks * 32, BLOCK, lane);
#pragma unroll
      for (int t = 0; t < 8; ++t) {
        v16bf bv = fragB_lds(lds_vt + t * 16 * BLOCK + ks * 32, BLOCK, lane);
        acc[t] = __builtin_amdgcn_wmma_f32_16x16x32_bf16(
                     false, ap, false, bv, (short)0, acc[t], false, false);
      }
    }
    __syncthreads();   // protect lds_k/lds_vt before next block's loads
  }

  // ---- store O back as f32 [B,S,hidden] ----
  const int row0 = qblk * BLOCK + wave * 16 + ((lane >> 4) << 3);
  const int col0 = h * D + (lane & 15);
#pragma unroll
  for (int t = 0; t < 8; ++t) {
#pragma unroll
    for (int j = 0; j < 8; ++j) {
      out[(size_t)(row0 + j) * HIDDEN + col0 + t * 16] = acc[t][j];
    }
  }
}

extern "C" void kernel_launch(void* const* d_in, const int* in_sizes, int n_in,
                              void* d_out, int out_size, void* d_ws, size_t ws_size,
                              hipStream_t stream) {
  const float* query   = (const float*)d_in[0];
  const float* key     = (const float*)d_in[1];
  const float* value   = (const float*)d_in[2];
  const float* cache_k = (const float*)d_in[3];
  const float* cache_v = (const float*)d_in[4];
  const int*   pos     = (const int*)d_in[5];
  float* out = (float*)d_out;

  const int S = in_sizes[0] / HIDDEN;        // 4096
  dim3 grid(H, S / BLOCK);                   // (16, 32)
  sparse_attn_bf16_wmma<<<grid, THREADS, 0, stream>>>(
      query, key, value, cache_k, cache_v, pos, out);
}
